// sigmaGPT_28887950033545
// MI455X (gfx1250) — matrix operand order repaired and checked by reading, not compiled, not run on hardware
//
#include <hip/hip_runtime.h>
#include <hip/hip_bf16.h>

// ---------------------------------------------------------------------------
// sigmaGPT attention block for gfx1250 (MI455X): bf16 WMMA + double-buffered
// async global->LDS staging.  B=8, T=512, C=1024, H=16, E=64, L=2048, S=2560
// ---------------------------------------------------------------------------

#define BB 8
#define TT 512
#define CC 1024
#define HH 16
#define EE 64
#define LL 2048
#define SS (LL + TT)

typedef __attribute__((ext_vector_type(16))) __bf16 v16bf;
typedef __attribute__((ext_vector_type(8)))  float  v8f;

union frag_u { v16bf v; unsigned d[8]; };

__device__ __forceinline__ unsigned short f2bf(float f) {
  union { float f; unsigned u; } x;
  x.f = f;
  unsigned r = x.u + 0x7FFFu + ((x.u >> 16) & 1u);   // round-to-nearest-even
  return (unsigned short)(r >> 16);
}

#if __has_builtin(__builtin_amdgcn_cvt_pk_bf16_f32)
typedef __attribute__((ext_vector_type(2))) __bf16 v2bf;
__device__ __forceinline__ unsigned packbf2(float a, float b) {
  union { v2bf v; unsigned u; } u;
  u.v = __builtin_amdgcn_cvt_pk_bf16_f32(a, b);
  return u.u;
}
#else
__device__ __forceinline__ unsigned packbf2(float a, float b) {
  return (unsigned)f2bf(a) | ((unsigned)f2bf(b) << 16);
}
#endif

__device__ __forceinline__ v8f wmma_bf16(const frag_u& a, const frag_u& b, v8f c) {
  return __builtin_amdgcn_wmma_f32_16x16x32_bf16(
      false, a.v, false, b.v, (short)0, c, false, false);
}

// async global->LDS copy of 8 bytes per lane (GVS mode), tracked on ASYNCcnt
__device__ __forceinline__ void async_b64(unsigned lds_addr, unsigned goff,
                                          unsigned long long base) {
  asm volatile("global_load_async_to_lds_b64 %0, %1, %2"
               :: "v"(lds_addr), "v"(goff), "s"(base) : "memory");
}
__device__ __forceinline__ void wait_async0() {
  asm volatile("s_wait_asynccnt 0" ::: "memory");
}
__device__ __forceinline__ void wait_async8() {   // oldest 8 of 16 complete
  asm volatile("s_wait_asynccnt 8" ::: "memory");
}

// ---------------------------------------------------------------------------
// GEMM: C[M,N] = A[M,K] @ W[K,N] + bias[N]  (f32 in/out, bf16 WMMA inner)
// 256 threads = 8 waves; wave w owns rows [m0+16w,16) x 64-wide column strip
// (A fragment reused across 4 WMMA per k-step). Global loads for k-step i+1
// are software-pipelined into registers during k-step i's WMMAs.
// Requires M%128==0, N%64==0, K%32==0.
// ---------------------------------------------------------------------------
__global__ void __launch_bounds__(256)
gemm_bf16(const float* __restrict__ A, const float* __restrict__ W,
          const float* __restrict__ bias, float* __restrict__ C,
          int M, int N, int K) {
  __shared__ unsigned short bT[64 * 40];        // [n][k], k pairs contiguous
  __shared__ unsigned short aT[8 * 16 * 40];    // per-wave [m][k]

  const int tid  = threadIdx.x;
  const int lane = tid & 31;
  const int wave = tid >> 5;
  const int half = lane >> 4;
  const int mn   = lane & 15;

  const int n0 = blockIdx.x * 64;
  const int m0 = blockIdx.y * 128 + wave * 16;

  unsigned short* aW = &aT[wave * 16 * 40];
  v8f acc0 = {}, acc1 = {}, acc2 = {}, acc3 = {};

  float  fB[4][2];
  float2 fA[8];

  // prologue: load k-step 0 into registers, convert+store to LDS
  #pragma unroll
  for (int i = 0; i < 4; ++i) {
    int idx = tid + 256 * i;
    int n = idx & 63, kp = idx >> 6;
    fB[i][0] = W[(size_t)(2 * kp) * N + n0 + n];
    fB[i][1] = W[(size_t)(2 * kp + 1) * N + n0 + n];
  }
  #pragma unroll
  for (int i = 0; i < 8; ++i) {
    int idx = lane + 32 * i;
    int m = idx >> 4, kp = idx & 15;
    fA[i] = *(const float2*)&A[(size_t)(m0 + m) * K + 2 * kp];
  }
  #pragma unroll
  for (int i = 0; i < 4; ++i) {
    int idx = tid + 256 * i;
    int n = idx & 63, kp = idx >> 6;
    *(unsigned*)&bT[n * 40 + 2 * kp] = packbf2(fB[i][0], fB[i][1]);
  }
  #pragma unroll
  for (int i = 0; i < 8; ++i) {
    int idx = lane + 32 * i;
    int m = idx >> 4, kp = idx & 15;
    *(unsigned*)&aW[m * 40 + 2 * kp] = packbf2(fA[i].x, fA[i].y);
  }
  __syncthreads();

  for (int kk = 0; kk < K; kk += 32) {
    const bool more = (kk + 32) < K;
    if (more) {                                  // prefetch next k-step -> regs
      #pragma unroll
      for (int i = 0; i < 4; ++i) {
        int idx = tid + 256 * i;
        int n = idx & 63, kp = idx >> 6;
        fB[i][0] = W[(size_t)(kk + 32 + 2 * kp) * N + n0 + n];
        fB[i][1] = W[(size_t)(kk + 32 + 2 * kp + 1) * N + n0 + n];
      }
      #pragma unroll
      for (int i = 0; i < 8; ++i) {
        int idx = lane + 32 * i;
        int m = idx >> 4, kp = idx & 15;
        fA[i] = *(const float2*)&A[(size_t)(m0 + m) * K + kk + 32 + 2 * kp];
      }
    }

    frag_u af, b0, b1, b2, b3;
    const unsigned short* ap = &aW[mn * 40];
    #pragma unroll
    for (int j = 0; j < 4; ++j) {
      af.d[j]     = *(const unsigned*)&ap[16 * half + 2 * j];
      af.d[j + 4] = *(const unsigned*)&ap[16 * half + 8 + 2 * j];
    }
    const unsigned short* bp0 = &bT[(mn)      * 40];
    const unsigned short* bp1 = &bT[(16 + mn) * 40];
    const unsigned short* bp2 = &bT[(32 + mn) * 40];
    const unsigned short* bp3 = &bT[(48 + mn) * 40];
    #pragma unroll
    for (int j = 0; j < 8; ++j) {
      b0.d[j] = *(const unsigned*)&bp0[16 * half + 2 * j];
      b1.d[j] = *(const unsigned*)&bp1[16 * half + 2 * j];
      b2.d[j] = *(const unsigned*)&bp2[16 * half + 2 * j];
      b3.d[j] = *(const unsigned*)&bp3[16 * half + 2 * j];
    }
    acc0 = wmma_bf16(af, b0, acc0);
    acc1 = wmma_bf16(af, b1, acc1);
    acc2 = wmma_bf16(af, b2, acc2);
    acc3 = wmma_bf16(af, b3, acc3);
    __syncthreads();                             // done reading LDS

    if (more) {                                  // convert+store next k-step
      #pragma unroll
      for (int i = 0; i < 4; ++i) {
        int idx = tid + 256 * i;
        int n = idx & 63, kp = idx >> 6;
        *(unsigned*)&bT[n * 40 + 2 * kp] = packbf2(fB[i][0], fB[i][1]);
      }
      #pragma unroll
      for (int i = 0; i < 8; ++i) {
        int idx = lane + 32 * i;
        int m = idx >> 4, kp = idx & 15;
        *(unsigned*)&aW[m * 40 + 2 * kp] = packbf2(fA[i].x, fA[i].y);
      }
      __syncthreads();
    }
  }

  const float bv0 = bias[n0 + mn];
  const float bv1 = bias[n0 + 16 + mn];
  const float bv2 = bias[n0 + 32 + mn];
  const float bv3 = bias[n0 + 48 + mn];
  #pragma unroll
  for (int r = 0; r < 8; ++r) {
    size_t row = (size_t)(m0 + 8 * half + r) * N + n0 + mn;
    C[row]      = acc0[r] + bv0;
    C[row + 16] = acc1[r] + bv1;
    C[row + 32] = acc2[r] + bv2;
    C[row + 48] = acc3[r] + bv3;
  }
}

// ---------------------------------------------------------------------------
// prep_kv: pack full K/V (cache + fresh rows from qkv ws) into bf16:
//   kbf (B,H,S,E) row-major, vbf (B,H,E,S) = V transposed (via LDS tile).
// ---------------------------------------------------------------------------
__global__ void __launch_bounds__(128)
prep_kv(const float* __restrict__ kc, const float* __restrict__ vc,
        const float* __restrict__ qkv,
        unsigned short* __restrict__ kbf, unsigned short* __restrict__ vbf) {
  __shared__ unsigned short tile[32 * 34];      // [e][s] staging for V^T

  const int tid = threadIdx.x;
  const int bh  = blockIdx.x;
  const int b   = bh / HH, h = bh % HH;
  const int s0  = blockIdx.y * 32;
  const int e0  = blockIdx.z * 32;
  const int hc  = h * EE;

  const float* kc_bh = kc + (size_t)bh * LL * EE;
  const float* vc_bh = vc + (size_t)bh * LL * EE;
  const float* qkv_b = qkv + (size_t)b * TT * 3 * CC;
  unsigned short* kb = kbf + (size_t)bh * SS * EE;
  unsigned short* vb = vbf + (size_t)bh * EE * SS;

  #pragma unroll
  for (int i = 0; i < 8; ++i) {
    int idx = tid + 128 * i;
    int el = idx & 31, sl = idx >> 5;
    int s = s0 + sl, e = e0 + el;
    float kvK, kvV;
    if (s < LL) {
      kvK = kc_bh[(size_t)s * EE + e];
      kvV = vc_bh[(size_t)s * EE + e];
    } else {
      const float* row = &qkv_b[(size_t)(s - LL) * 3 * CC];
      kvK = row[CC + hc + e];
      kvV = row[2 * CC + hc + e];
    }
    kb[(size_t)s * EE + e] = f2bf(kvK);
    tile[el * 34 + sl]     = f2bf(kvV);
  }
  __syncthreads();
  #pragma unroll
  for (int i = 0; i < 4; ++i) {
    int idx = tid + 128 * i;
    int el = idx >> 3, sp = idx & 7;
    *(unsigned*)&vb[(size_t)(e0 + el) * SS + s0 + 2 * sp] =
        *(const unsigned*)&tile[el * 34 + 2 * sp];
  }
}

// ---------------------------------------------------------------------------
// Flash attention with KV cache. Block = 128 threads (4 waves) = 64 query
// rows of one (b,h). 32-key chunks double-buffered with async global->LDS
// copies (split s_wait_asynccnt), online softmax in exp2 domain,
// 8 bf16 WMMA per chunk per wave.
// ---------------------------------------------------------------------------
__global__ void __launch_bounds__(128)
attn(const float* __restrict__ qkv,
     const unsigned short* __restrict__ kbf,
     const unsigned short* __restrict__ vbf,
     float* __restrict__ y) {
  __shared__ unsigned short Klds[2 * 32 * 72];   // ping-pong [key][e]
  __shared__ unsigned short Vlds[2 * 64 * 40];   // ping-pong [e][key]
  __shared__ unsigned short Qlds[4 * 16 * 72];   // per-wave [m][e]
  __shared__ unsigned short Plds[4 * 16 * 40];   // per-wave [m][key]

  const int tid  = threadIdx.x;
  const int lane = tid & 31;
  const int wave = tid >> 5;
  const int half = lane >> 4;
  const int mn   = lane & 15;

  const int bh = blockIdx.x;
  const int b  = bh / HH, h = bh % HH;
  const int qb_blk = blockIdx.y * 64;
  const int qb     = qb_blk + wave * 16;

  const float* qkv_b = qkv + (size_t)b * TT * 3 * CC;
  const int hc = h * EE;
  const unsigned short* kb_bh = kbf + (size_t)bh * SS * EE;
  const unsigned short* vb_bh = vbf + (size_t)bh * EE * SS;
  const unsigned long long kgb = (unsigned long long)(size_t)kb_bh;
  const unsigned long long vgb = (unsigned long long)(size_t)vb_bh;
  const unsigned kldsb = (unsigned)(size_t)&Klds[0];
  const unsigned vldsb = (unsigned)(size_t)&Vlds[0];

  // issue one chunk's K+V async copies into buffer `buf` (8 b64 per wave)
  auto stage_async = [&](int s0, int buf) {
    unsigned kb0 = kldsb + (unsigned)buf * (32 * 144);
    unsigned vb0 = vldsb + (unsigned)buf * (64 * 80);
    #pragma unroll
    for (int i = 0; i < 4; ++i) {
      int idx = tid + 128 * i;
      int key = idx >> 4, q = idx & 15;                     // K: 32x64 bf16
      async_b64(kb0 + key * 144 + q * 8,
                (unsigned)((((s0 + key) * EE) + q * 4) * 2), kgb);
      int e = idx >> 3, q2 = idx & 7;                       // V^T: 64x32
      async_b64(vb0 + e * 80 + q2 * 8,
                (unsigned)(((size_t)e * SS + s0 + q2 * 4) * 2), vgb);
    }
  };

  // ---- stage Q (16x64), pre-scaled by log2(e)/sqrt(E) for exp2 softmax ----
  const float QSCALE = 0.125f * 1.44269504088896340736f;
  unsigned short* qW = &Qlds[wave * 16 * 72];
  #pragma unroll
  for (int m = 0; m < 16; ++m) {
    const float* qrow = &qkv_b[(size_t)(qb + m) * 3 * CC + hc];
    qW[m * 72 + lane]      = f2bf(qrow[lane] * QSCALE);
    qW[m * 72 + 32 + lane] = f2bf(qrow[32 + lane] * QSCALE);
  }

  stage_async(0, 0);                             // prologue: chunk 0 -> buf 0
  __syncthreads();

  frag_u a0, a1;
  {
    const unsigned short* qp = &qW[mn * 72];
    #pragma unroll
    for (int j = 0; j < 4; ++j) {
      a0.d[j]     = *(const unsigned*)&qp[16 * half + 2 * j];
      a0.d[j + 4] = *(const unsigned*)&qp[16 * half + 8 + 2 * j];
      a1.d[j]     = *(const unsigned*)&qp[32 + 16 * half + 2 * j];
      a1.d[j + 4] = *(const unsigned*)&qp[32 + 16 * half + 8 + 2 * j];
    }
  }

  v8f ot0 = {}, ot1 = {}, ot2 = {}, ot3 = {};
  float Mr[8], Lr[8];
  #pragma unroll
  for (int r = 0; r < 8; ++r) { Mr[r] = -3.0e38f; Lr[r] = 0.0f; }

  unsigned short* pW = &Plds[wave * 16 * 40];
  const int send = LL + qb_blk + 64;             // multiple of 32

  for (int s0 = 0; s0 < send; s0 += 32) {
    const int cur = (s0 >> 5) & 1;
    if (s0 + 32 < send) {                        // stream next chunk behind us
      stage_async(s0 + 32, cur ^ 1);
      wait_async8();                             // oldest 8 (chunk s0) landed
    } else {
      wait_async0();
    }
    __syncthreads();

    const unsigned short* Kbuf = &Klds[cur * 32 * 72];
    const unsigned short* Vbuf = &Vlds[cur * 64 * 40];

    // ---- S = Q @ K^T (two 16-key column tiles) ----
    v8f c0 = {}, c1 = {};
    {
      frag_u b00, b01, b10, b11;
      const unsigned short* kp0 = &Kbuf[mn * 72];
      const unsigned short* kp1 = &Kbuf[(16 + mn) * 72];
      #pragma unroll
      for (int j = 0; j < 8; ++j) {
        b00.d[j] = *(const unsigned*)&kp0[16 * half + 2 * j];
        b01.d[j] = *(const unsigned*)&kp0[32 + 16 * half + 2 * j];
        b10.d[j] = *(const unsigned*)&kp1[16 * half + 2 * j];
        b11.d[j] = *(const unsigned*)&kp1[32 + 16 * half + 2 * j];
      }
      c0 = wmma_bf16(a0, b00, c0);
      c0 = wmma_bf16(a1, b01, c0);
      c1 = wmma_bf16(a0, b10, c1);
      c1 = wmma_bf16(a1, b11, c1);
    }

    // ---- causal mask only on diagonal-crossing chunks ----
    if (s0 + 31 > LL + qb) {
      int lim = LL + qb + 8 * half;
      int sA = s0 + mn, sB = s0 + 16 + mn;
      #pragma unroll
      for (int r = 0; r < 8; ++r) {
        if (sA > lim + r) c0[r] = -3.0e38f;
        if (sB > lim + r) c1[r] = -3.0e38f;
      }
    }

    // ---- online softmax in exp2 domain (row r lives in a 16-lane half) ----
    #pragma unroll
    for (int r = 0; r < 8; ++r) {
      float mx = fmaxf(c0[r], c1[r]);
      #pragma unroll
      for (int off = 1; off < 16; off <<= 1)
        mx = fmaxf(mx, __shfl_xor(mx, off, 32));
      float Mn = fmaxf(Mr[r], mx);
      float alpha = __builtin_amdgcn_exp2f(Mr[r] - Mn);
      float p0 = __builtin_amdgcn_exp2f(c0[r] - Mn);
      float p1 = __builtin_amdgcn_exp2f(c1[r] - Mn);
      float rs = p0 + p1;
      #pragma unroll
      for (int off = 1; off < 16; off <<= 1)
        rs += __shfl_xor(rs, off, 32);
      Lr[r] = Lr[r] * alpha + rs;
      Mr[r] = Mn;
      pW[(8 * half + r) * 40 + mn]      = f2bf(p0);
      pW[(8 * half + r) * 40 + 16 + mn] = f2bf(p1);
      ot0[r] *= alpha; ot1[r] *= alpha;
      ot2[r] *= alpha; ot3[r] *= alpha;
    }

    // ---- O += P @ V ----
    {
      frag_u pa;
      const unsigned short* pp = &pW[mn * 40];
      #pragma unroll
      for (int j = 0; j < 4; ++j) {
        pa.d[j]     = *(const unsigned*)&pp[16 * half + 2 * j];
        pa.d[j + 4] = *(const unsigned*)&pp[16 * half + 8 + 2 * j];
      }
      frag_u bv0, bv1, bv2, bv3;
      const unsigned short* vp0 = &Vbuf[(mn)      * 40];
      const unsigned short* vp1 = &Vbuf[(16 + mn) * 40];
      const unsigned short* vp2 = &Vbuf[(32 + mn) * 40];
      const unsigned short* vp3 = &Vbuf[(48 + mn) * 40];
      #pragma unroll
      for (int j = 0; j < 8; ++j) {
        bv0.d[j] = *(const unsigned*)&vp0[16 * half + 2 * j];
        bv1.d[j] = *(const unsigned*)&vp1[16 * half + 2 * j];
        bv2.d[j] = *(const unsigned*)&vp2[16 * half + 2 * j];
        bv3.d[j] = *(const unsigned*)&vp3[16 * half + 2 * j];
      }
      ot0 = wmma_bf16(pa, bv0, ot0);
      ot1 = wmma_bf16(pa, bv1, ot1);
      ot2 = wmma_bf16(pa, bv2, ot2);
      ot3 = wmma_bf16(pa, bv3, ot3);
    }
    __syncthreads();                             // reads done before re-stage
  }

  // ---- epilogue: normalize and write y (B,T,C) ----
  #pragma unroll
  for (int r = 0; r < 8; ++r) {
    float inv = __builtin_amdgcn_rcpf(Lr[r]);
    size_t row = (size_t)(b * TT + qb + 8 * half + r) * CC + hc;
    y[row + mn]      = ot0[r] * inv;
    y[row + 16 + mn] = ot1[r] * inv;
    y[row + 32 + mn] = ot2[r] * inv;
    y[row + 48 + mn] = ot3[r] * inv;
  }
}

// ---------------------------------------------------------------------------
extern "C" void kernel_launch(void* const* d_in, const int* in_sizes, int n_in,
                              void* d_out, int out_size, void* d_ws, size_t ws_size,
                              hipStream_t stream) {
  const float* x       = (const float*)d_in[0];
  const float* k_cache = (const float*)d_in[1];
  const float* v_cache = (const float*)d_in[2];
  const float* Wqkv    = (const float*)d_in[3];
  const float* bqkv    = (const float*)d_in[4];
  const float* Wproj   = (const float*)d_in[5];
  const float* bproj   = (const float*)d_in[6];
  float* out = (float*)d_out;

  float* qkv_ws = (float*)d_ws;                                  // (B,T,3C) f32
  float* y_ws   = qkv_ws + (size_t)BB * TT * 3 * CC;             // (B,T,C)  f32
  unsigned short* kbf = (unsigned short*)(y_ws + (size_t)BB * TT * CC); // (B,H,S,E)
  unsigned short* vbf = kbf + (size_t)BB * HH * SS * EE;         // (B,H,E,S)

  const int M = BB * TT;                                         // 4096

  // 1) qkv = x @ Wqkv + bqkv
  gemm_bf16<<<dim3((3 * CC) / 64, M / 128), 256, 0, stream>>>(
      x, Wqkv, bqkv, qkv_ws, M, 3 * CC, CC);

  // 2) pack K/V (cache + fresh) to bf16; V transposed for PV B-fragments
  prep_kv<<<dim3(BB * HH, SS / 32, EE / 32), 128, 0, stream>>>(
      k_cache, v_cache, qkv_ws, kbf, vbf);

  // 3) flash attention (double-buffered async-LDS, bf16 WMMA)
  attn<<<dim3(BB * HH, TT / 64), 128, 0, stream>>>(qkv_ws, kbf, vbf, y_ws);

  // 4) out = y @ Wproj + bproj
  gemm_bf16<<<dim3(CC / 64, M / 128), 256, 0, stream>>>(
      y_ws, Wproj, bproj, out, M, CC, CC);
}
